// Non_Local_Block_1709396984536
// MI455X (gfx1250) — compile-verified, matrix operand order checked by reading, not run on hardware
//
#include <hip/hip_runtime.h>
#include <hip/hip_bf16.h>

// ---------------------------------------------------------------------------
// Non-local block, fused pipeline for gfx1250 (MI455X), wave32 + WMMA bf16.
//   K1 proj : thetaT/phiT ([B][L][M] bf16) and gamma ([B][M][L] bf16),
//             software-pipelined (next tile loads overlap current WMMA)
//   K2 attn : flash-style fused scores+softmax+PV -> yT ([B][L][M] bf16),
//             double-buffered GLOBAL_LOAD_ASYNC_TO_LDS staging (ASYNCcnt)
//   K3 out  : out = x + lrelu(W_o @ y + b_o), fp32; yT tile async-staged
// ---------------------------------------------------------------------------

typedef __attribute__((ext_vector_type(16))) __bf16 v16bf;
typedef __attribute__((ext_vector_type(8)))  __bf16 v8bf;
typedef __attribute__((ext_vector_type(8)))  float  v8f;
typedef int v4i_vs __attribute__((vector_size(16)));   // <4 x i32>

constexpr int Bn = 4;
constexpr int Cc = 2048;
constexpr int Mm = 256;
constexpr int Ll = 4096;

union frag16 { v16bf v; v8bf h[2]; };

__device__ __forceinline__ float lrelu(float v) { return v >= 0.f ? v : 0.2f * v; }

__device__ __forceinline__ v8f wmma_bf16(const frag16& a, const frag16& b, v8f c) {
    return __builtin_amdgcn_wmma_f32_16x16x32_bf16(
        /*neg_a=*/false, a.v, /*neg_b=*/false, b.v,
        /*c_mod=*/(short)0, c, /*reuse_a=*/false, /*reuse_b=*/false);
}

// Gather a bf16 fragment whose per-lane data is contiguous in K:
// elements 0..7  <- src[half*8 .. +7], elements 8..15 <- src[16+half*8 .. +7]
__device__ __forceinline__ frag16 load_frag(const __bf16* src, int half) {
    frag16 f;
    f.h[0] = *(const v8bf*)(src + half * 8);
    f.h[1] = *(const v8bf*)(src + 16 + half * 8);
    return f;
}

// ---- CDNA5 async global->LDS copy (ASYNCcnt path), with safe fallback ------
#if defined(__has_builtin)
#  if __has_builtin(__builtin_amdgcn_global_load_async_to_lds_b128)
#    define HAVE_ASYNC_COPY 1
#  endif
#endif
#ifndef HAVE_ASYNC_COPY
#  define HAVE_ASYNC_COPY 0
#endif

__device__ __forceinline__ void copy16_to_lds(void* lds, const void* glob) {
#if HAVE_ASYNC_COPY
    __builtin_amdgcn_global_load_async_to_lds_b128(
        (__attribute__((address_space(1))) v4i_vs*)glob,
        (__attribute__((address_space(3))) v4i_vs*)lds,
        /*offset=*/0, /*cpol=*/0);
#else
    *(float4*)lds = *(const float4*)glob;   // 16B global load + 16B ds store
#endif
}

__device__ __forceinline__ void wait_async0() {
#if defined(__has_builtin) && __has_builtin(__builtin_amdgcn_s_wait_asynccnt)
    __builtin_amdgcn_s_wait_asynccnt(0);
#else
    asm volatile("s_wait_asynccnt 0x0" ::: "memory");
#endif
}

// ---------------------------------------------------------------------------
// K1: projections.  A = W (row-major, K=C contiguous), B = x tile transposed
// through LDS with fp32->bf16 conversion.  Software-pipelined: tile k+1 is
// loaded into registers while tile k's WMMAs run.  Workgroup = 128 thr,
// 64(M) x 64(L) output tile.  grid = (L/64, 3*M/64, B)
// ---------------------------------------------------------------------------
__global__ void __launch_bounds__(128)
proj_kernel(const float* __restrict__ x,
            const float* __restrict__ Wt, const float* __restrict__ bt,
            const float* __restrict__ Wp, const float* __restrict__ bp,
            const float* __restrict__ Wg, const float* __restrict__ bg,
            __bf16* __restrict__ thetaT, __bf16* __restrict__ phiT,
            __bf16* __restrict__ gamma)
{
    __shared__ __align__(16) __bf16 As[64 * 32];   // [m][k]
    __shared__ __align__(16) __bf16 BsT[64 * 32];  // [l][k] (x tile transposed)

    const int tid = threadIdx.x;
    const int lane = tid & 31, wave = tid >> 5;
    const int half = lane >> 4, n16 = lane & 15;

    const int b  = blockIdx.z;
    const int l0 = blockIdx.x * 64;
    const int gm0 = blockIdx.y * 64;   // 0..704 over stacked [Wt;Wp;Wg]
    const int sec = gm0 >> 8;          // 0=theta 1=phi 2=gamma
    const int m0  = gm0 & 255;         // row base within section

    const float* W    = (sec == 0) ? Wt : (sec == 1) ? Wp : Wg;
    const float* bias = (sec == 0) ? bt : (sec == 1) ? bp : bg;
    const float* xb   = x + (size_t)b * Cc * Ll;

    v8f acc[4];
    for (int t = 0; t < 4; ++t)
        for (int i = 0; i < 8; ++i) acc[t][i] = 0.f;

    float4 aReg[4], bReg[4];
    auto load_tile = [&](int c0) {
        for (int i = 0; i < 4; ++i) {
            const int q = tid + i * 128;
            aReg[i] = *(const float4*)(W  + (size_t)(m0 + (q >> 3)) * Cc + c0 + (q & 7) * 4);
            bReg[i] = *(const float4*)(xb + (size_t)(c0 + (q >> 4)) * Ll + l0 + (q & 15) * 4);
        }
    };

    load_tile(0);
    for (int c0 = 0; c0 < Cc; c0 += 32) {
        // commit staged registers -> LDS (fp32 -> bf16, B transposed)
        for (int i = 0; i < 4; ++i) {
            const int q = tid + i * 128;
            {
                const int r = q >> 3, c4 = (q & 7) * 4;
                __bf16* d = &As[r * 32 + c4];
                d[0] = (__bf16)aReg[i].x; d[1] = (__bf16)aReg[i].y;
                d[2] = (__bf16)aReg[i].z; d[3] = (__bf16)aReg[i].w;
            }
            {
                const int r = q >> 4, c4 = (q & 15) * 4;
                BsT[(c4 + 0) * 32 + r] = (__bf16)bReg[i].x;
                BsT[(c4 + 1) * 32 + r] = (__bf16)bReg[i].y;
                BsT[(c4 + 2) * 32 + r] = (__bf16)bReg[i].z;
                BsT[(c4 + 3) * 32 + r] = (__bf16)bReg[i].w;
            }
        }
        __syncthreads();

        if (c0 + 32 < Cc) load_tile(c0 + 32);   // overlaps with WMMA below
        if (c0 + 64 < Cc)                       // speculative prefetch 2 tiles ahead
            __builtin_prefetch(xb + (size_t)(c0 + 64 + (tid >> 2)) * Ll + l0 + (tid & 3) * 16, 0, 1);

        const frag16 a = load_frag(&As[(wave * 16 + n16) * 32], half);
        for (int nt = 0; nt < 4; ++nt) {
            const frag16 bf_ = load_frag(&BsT[(nt * 16 + n16) * 32], half);
            acc[nt] = wmma_bf16(a, bf_, acc[nt]);
        }
        __syncthreads();
    }

    // epilogue: bias + lrelu + bf16 store
    const int mbase = m0 + wave * 16 + half * 8;   // row base within section
    for (int nt = 0; nt < 4; ++nt) {
        const int l = l0 + nt * 16 + n16;
        if (sec < 2) {
            // transposed store: 8 consecutive m per lane -> one 16B store
            v8bf pack;
            for (int r = 0; r < 8; ++r)
                pack[r] = (__bf16)lrelu(acc[nt][r] + bias[mbase + r]);
            __bf16* dst = (sec == 0 ? thetaT : phiT) + ((size_t)b * Ll + l) * Mm + mbase;
            *(v8bf*)dst = pack;
        } else {
            for (int r = 0; r < 8; ++r) {
                const int ml = mbase + r;
                gamma[((size_t)b * Mm + ml) * Ll + l] =
                    (__bf16)lrelu(acc[nt][r] + bias[ml]);
            }
        }
    }
}

// ---------------------------------------------------------------------------
// K2: fused attention (flash-style), double-buffered async staging.
// Workgroup = 128 thr; each wave owns 16 query rows, streams over 64-wide
// key blocks.  While block j is computed, block j+1's thetaT/gamma tiles are
// in flight via GLOBAL_LOAD_ASYNC_TO_LDS; the only wait is at iteration end.
// grid = (L/64, B)
// ---------------------------------------------------------------------------
__global__ void __launch_bounds__(128)
attn_kernel(const __bf16* __restrict__ thetaT,
            const __bf16* __restrict__ phiT,
            const __bf16* __restrict__ gamma,
            __bf16* __restrict__ yT)
{
    __shared__ __align__(16) __bf16 Ts[2][64 * 256];  // thetaT tiles [j][m], 2x32KB
    __shared__ __align__(16) __bf16 Gs[2][256 * 64];  // gamma  tiles [m][j], 2x32KB
    __shared__ __align__(16) __bf16 Ps[4][16][64];    // per-wave P tile,        8KB

    const int tid = threadIdx.x;
    const int lane = tid & 31, wave = tid >> 5;
    const int half = lane >> 4, n16 = lane & 15;

    const int b  = blockIdx.y;
    const int i0 = blockIdx.x * 64 + wave * 16;

    const __bf16* phiRow = phiT + ((size_t)b * Ll + i0 + n16) * Mm;  // A rows (K=M)
    const __bf16* thB    = thetaT + (size_t)b * Ll * Mm;
    const __bf16* gmB    = gamma  + (size_t)b * Mm * Ll;

    auto stage_tiles = [&](int buf, int j0) {
        // thetaT[j0..+64][0..256): flat contiguous 32KB
        const char* tsrc = (const char*)(thB + (size_t)j0 * Mm);
        char* tdst = (char*)Ts[buf];
        for (int q = tid; q < 2048; q += 128)
            copy16_to_lds(tdst + q * 16, tsrc + q * 16);
        // gamma[0..256)[j0..+64): 256 rows x 128B
        const char* gsrc = (const char*)(gmB + j0);
        char* gdst = (char*)Gs[buf];
        for (int q = tid; q < 2048; q += 128) {
            const int m = q >> 3, o = (q & 7) * 16;
            copy16_to_lds(gdst + m * 128 + o, gsrc + (size_t)m * Ll * 2 + o);
        }
    };

    v8f yacc[16];
    for (int mt = 0; mt < 16; ++mt)
        for (int r = 0; r < 8; ++r) yacc[mt][r] = 0.f;
    float mstat[8], lstat[8];
    for (int r = 0; r < 8; ++r) { mstat[r] = -1e30f; lstat[r] = 0.f; }

    constexpr int NB = Ll / 64;
    stage_tiles(0, 0);
    wait_async0();
    __syncthreads();

    for (int jb = 0; jb < NB; ++jb) {
        const int cur = jb & 1;
        // kick off next block's tiles; lands while we compute this block.
        // (waves finished reading Ts/Gs[cur^1] at the barrier ending jb-1)
        if (jb + 1 < NB) stage_tiles(cur ^ 1, (jb + 1) * 64);

        const __bf16* Tsb = Ts[cur];
        const __bf16* Gsb = Gs[cur];

        // ---- S = phi^T @ theta for this 16i x 64j strip (K = 256) ----
        v8f s[4];
        for (int nt = 0; nt < 4; ++nt)
            for (int r = 0; r < 8; ++r) s[nt][r] = 0.f;
        for (int kk = 0; kk < 8; ++kk) {
            const int k0 = kk * 32;
            const frag16 a = load_frag(phiRow + k0, half);
            for (int nt = 0; nt < 4; ++nt) {
                const frag16 bf_ = load_frag(&Tsb[(nt * 16 + n16) * Mm + k0], half);
                s[nt] = wmma_bf16(a, bf_, s[nt]);
            }
        }
        for (int nt = 0; nt < 4; ++nt)
            for (int r = 0; r < 8; ++r) s[nt][r] *= 0.03125f;   // 1/32 scale

        // ---- online softmax (rows live across 16 lanes of each half) ----
        for (int r = 0; r < 8; ++r) {
            float v = fmaxf(fmaxf(s[0][r], s[1][r]), fmaxf(s[2][r], s[3][r]));
            for (int d = 1; d < 16; d <<= 1) v = fmaxf(v, __shfl_xor(v, d, 32));
            const float mnew = fmaxf(mstat[r], v);
            const float corr = __expf(mstat[r] - mnew);
            mstat[r] = mnew;
            lstat[r] *= corr;
            for (int mt = 0; mt < 16; ++mt) yacc[mt][r] *= corr;
            float sum = 0.f;
            for (int nt = 0; nt < 4; ++nt) {
                const float p = __expf(s[nt][r] - mnew);
                s[nt][r] = p;
                sum += p;
            }
            for (int d = 1; d < 16; d <<= 1) sum += __shfl_xor(sum, d, 32);
            lstat[r] += sum;
        }

        // ---- transpose P through per-wave LDS (C-layout -> A-layout) ----
        for (int nt = 0; nt < 4; ++nt)
            for (int r = 0; r < 8; ++r)
                Ps[wave][r + half * 8][nt * 16 + n16] = (__bf16)s[nt][r];
        // per-wave private region: in-wave DS ordering suffices, no barrier

        // ---- Y += P @ gamma^T  (K = 64 j, N = 256 m) ----
        for (int ks = 0; ks < 2; ++ks) {
            const frag16 a = load_frag(&Ps[wave][n16][ks * 32], half);
            for (int mt = 0; mt < 16; ++mt) {
                const frag16 bf_ = load_frag(&Gsb[(mt * 16 + n16) * 64 + ks * 32], half);
                yacc[mt] = wmma_bf16(a, bf_, yacc[mt]);
            }
        }

        wait_async0();     // next block's tiles have landed
        __syncthreads();   // publish them / retire this block's buffers
    }

    // ---- normalize and store yT[b][i][m] (bf16) ----
    for (int r = 0; r < 8; ++r) {
        const float inv = 1.f / lstat[r];
        const int i = i0 + r + half * 8;
        for (int mt = 0; mt < 16; ++mt)
            yT[((size_t)b * Ll + i) * Mm + mt * 16 + n16] =
                (__bf16)(yacc[mt][r] * inv);
    }
}

// ---------------------------------------------------------------------------
// K3: out = x + lrelu(W_o @ y + b_o).  A = W_o rows (fp32->bf16 on the fly,
// K=M contiguous), B = yT tile [64l x 256m] async-staged once to LDS and
// shared by all 4 waves.  grid = (L/64, C/64, B)
// ---------------------------------------------------------------------------
__global__ void __launch_bounds__(128)
out_kernel(const float* __restrict__ x, const float* __restrict__ Wo,
           const float* __restrict__ bo, const __bf16* __restrict__ yT,
           float* __restrict__ out)
{
    __shared__ __align__(16) __bf16 Ys[64 * 256];   // yT tile [l][m], 32KB

    const int tid = threadIdx.x;
    const int lane = tid & 31, wave = tid >> 5;
    const int half = lane >> 4, n16 = lane & 15;

    const int b  = blockIdx.z;
    const int l0 = blockIdx.x * 64;
    const int c0 = blockIdx.y * 64 + wave * 16;

    const float*  wr = Wo + (size_t)(c0 + n16) * Mm;   // A row for this lane
    const __bf16* yB = yT + (size_t)b * Ll * Mm;

    // async-stage yT[l0..+64][0..256) -> Ys (flat contiguous 32KB)
    {
        const char* ysrc = (const char*)(yB + (size_t)l0 * Mm);
        char* ydst = (char*)Ys;
        for (int q = tid; q < 2048; q += 128)
            copy16_to_lds(ydst + q * 16, ysrc + q * 16);
        wait_async0();
        __syncthreads();
    }

    v8f acc[4];
    for (int t = 0; t < 4; ++t)
        for (int i = 0; i < 8; ++i) acc[t][i] = 0.f;

    for (int kk = 0; kk < 8; ++kk) {
        const int k0 = kk * 32;
        float tmp[16];
        *(float4*)(tmp + 0)  = *(const float4*)(wr + k0 + half * 8);
        *(float4*)(tmp + 4)  = *(const float4*)(wr + k0 + half * 8 + 4);
        *(float4*)(tmp + 8)  = *(const float4*)(wr + k0 + 16 + half * 8);
        *(float4*)(tmp + 12) = *(const float4*)(wr + k0 + 16 + half * 8 + 4);
        frag16 a;
        for (int e = 0; e < 16; ++e) a.v[e] = (__bf16)tmp[e];

        for (int nt = 0; nt < 4; ++nt) {
            const frag16 bf_ = load_frag(&Ys[(nt * 16 + n16) * Mm + k0], half);
            acc[nt] = wmma_bf16(a, bf_, acc[nt]);
        }
    }

    for (int nt = 0; nt < 4; ++nt) {
        const int l = l0 + nt * 16 + n16;
        for (int r = 0; r < 8; ++r) {
            const int c = c0 + r + half * 8;
            const size_t idx = ((size_t)b * Cc + c) * Ll + l;
            out[idx] = x[idx] + lrelu(acc[nt][r] + bo[c]);
        }
    }
}

// ---------------------------------------------------------------------------
extern "C" void kernel_launch(void* const* d_in, const int* in_sizes, int n_in,
                              void* d_out, int out_size, void* d_ws, size_t ws_size,
                              hipStream_t stream) {
    (void)in_sizes; (void)n_in; (void)out_size; (void)ws_size;

    const float* x  = (const float*)d_in[0];
    const float* Wt = (const float*)d_in[1];
    const float* bt = (const float*)d_in[2];
    const float* Wp = (const float*)d_in[3];
    const float* bp = (const float*)d_in[4];
    const float* Wg = (const float*)d_in[5];
    const float* bg = (const float*)d_in[6];
    const float* Wo = (const float*)d_in[7];
    const float* bo = (const float*)d_in[8];
    float* out = (float*)d_out;

    // workspace layout (bf16): thetaT | phiT | gamma | yT  (8 MB each, 32 MB)
    const size_t SEC = (size_t)Bn * Ll * Mm;            // elements per tensor
    char* ws = (char*)d_ws;
    __bf16* thetaT = (__bf16*)ws;
    __bf16* phiT   = thetaT + SEC;
    __bf16* gamma  = phiT + SEC;
    __bf16* yT     = gamma + SEC;

    dim3 g1(Ll / 64, (3 * Mm) / 64, Bn);
    proj_kernel<<<g1, 128, 0, stream>>>(x, Wt, bt, Wp, bp, Wg, bg,
                                        thetaT, phiT, gamma);

    dim3 g2(Ll / 64, Bn);
    attn_kernel<<<g2, 128, 0, stream>>>(thetaT, phiT, gamma, yT);

    dim3 g3(Ll / 64, Cc / 64, Bn);
    out_kernel<<<g3, 128, 0, stream>>>(x, Wo, bo, yT, out);
}
